// SSGClassification_55972013801666
// MI455X (gfx1250) — matrix operand order repaired and checked by reading, not compiled, not run on hardware
//
#include <hip/hip_runtime.h>
#include <cstdint>

// ---------------- problem constants ----------------
#define BATCH   16
#define NPTS    8192
#define S1      512
#define S2      128
#define KNN     64        // k=50 padded to 64 (pad = duplicate first neighbor; max-pool invariant)
#define KRAW    50
#define NCATS   40
#define CATPAD  48
#define EPSV    1e-5f
#define FPS_T   256

// ---------------- CDNA5 WMMA types ----------------
typedef __attribute__((ext_vector_type(16))) __bf16 bf16x16;
typedef __attribute__((ext_vector_type(8)))  float  f32x8;

union Frag32B { uint4 q[2]; bf16x16 v; };   // 32 bytes = one wave32 bf16 fragment per lane

__device__ __forceinline__ __bf16 f2bf(float f) { return (__bf16)f; }
__device__ __forceinline__ float  bf2f(__bf16 h) { return (float)h; }

// ---------------------------------------------------------------------------
// bf16 WMMA GEMM with register blocking over N:  out = act(A[M][Kpad] x Bp + bias)
//   A   : row-major bf16, Kpad multiple of 32
//   Bp  : pre-swizzled weights; per (ktile t, col n, half h) a contiguous
//         32-byte chunk holding B[k][n] for k = 32t + 16h + 0..15
//   Each wave owns one 16-row M-tile and NT consecutive 16-col N-tiles,
//   reusing its A fragment NT times per k-step (NT wmmas / (2+2*NT) b128 loads).
//   Block = WPB waves (WPB M-tiles); grid = (M/16/WPB, Npad/16/NT)
// ---------------------------------------------------------------------------
template <int NT>
__global__ __launch_bounds__(128)
void wmma_gemm_kernel(const __bf16* __restrict__ A, const __bf16* __restrict__ Bp,
                      const float* __restrict__ bias,
                      __bf16* __restrict__ outB, float* __restrict__ outF,
                      int Kpad, int Npad, int relu)
{
  const int wave = threadIdx.x >> 5;
  const int wpb  = blockDim.x >> 5;
  const int lane = threadIdx.x & 31;
  const int lid  = lane & 15;
  const int hi   = lane >> 4;                  // K-half select (this code's contiguous k = 16*hi + s order for A and B)
  const size_t m0 = ((size_t)blockIdx.x * wpb + wave) * 16;
  const int    n0 = blockIdx.y * (16 * NT);

  f32x8 acc[NT] = {};
  const __bf16* arow  = A + (m0 + lid) * (size_t)Kpad + hi * 16;
  const __bf16* bcol0 = Bp + (((size_t)(n0 + lid)) * 2 + hi) * 16;
  const int ktiles = Kpad >> 5;
  for (int t = 0; t < ktiles; ++t) {
    Frag32B a;
    const __bf16* ap = arow + t * 32;
    a.q[0] = *(const uint4*)(ap);              // k = 32t + 16hi + 0..7
    a.q[1] = *(const uint4*)(ap + 8);          // k = 32t + 16hi + 8..15
    __builtin_prefetch((const void*)(ap + 32), 0, 1);   // global_prefetch_b8 next k-tile

    Frag32B b[NT];
    const __bf16* bp = bcol0 + (size_t)t * Npad * 32;
#pragma unroll
    for (int nt = 0; nt < NT; ++nt) {
      b[nt].q[0] = *(const uint4*)(bp + nt * 512);      // 16 cols = 512 bf16 apart
      b[nt].q[1] = *(const uint4*)(bp + nt * 512 + 8);
    }
#pragma unroll
    for (int nt = 0; nt < NT; ++nt)
      acc[nt] = __builtin_amdgcn_wmma_f32_16x16x32_bf16(false, a.v, false, b[nt].v,
                                                        (short)0, acc[nt], false, false);
  }

#pragma unroll
  for (int nt = 0; nt < NT; ++nt) {
    const int   n  = n0 + nt * 16 + lid;
    const float bv = bias ? bias[n] : 0.0f;
#pragma unroll
    for (int i = 0; i < 8; ++i) {
      size_t m = m0 + i + hi * 8;
      float v = acc[nt][i] + bv;
      if (relu) v = v > 0.0f ? v : 0.0f;
      if (outB) outB[m * (size_t)Npad + n] = f2bf(v);
      if (outF) outF[m * (size_t)Npad + n] = v;
    }
  }
}

// ---------------------------------------------------------------------------
// Fold BN into weights, pad to (Kpad,Npad), emit fragment-swizzled bf16 + bias
// ---------------------------------------------------------------------------
__global__ void prep_weights_kernel(const float* __restrict__ w, const float* __restrict__ b,
                                    const float* __restrict__ g, const float* __restrict__ be,
                                    __bf16* __restrict__ Bp, float* __restrict__ bias,
                                    int cout, int cin, int Kpad, int Npad)
{
  int idx = blockIdx.x * blockDim.x + threadIdx.x;
  if (idx >= Kpad * Npad) return;
  int k = idx / Npad, n = idx % Npad;
  float scale = 1.0f, shift = 0.0f;
  if (n < cout) {
    if (g)  scale = g[n] * rsqrtf(1.0f + EPSV);
    if (be) shift = be[n];
  }
  float v = (k < cin && n < cout) ? w[(size_t)n * cin + k] * scale : 0.0f;
  int t = k >> 5, kin = k & 31, h = kin >> 4;
  Bp[(((size_t)t * Npad + n) * 2 + h) * 16 + (kin & 15)] = f2bf(v);
  if (k == 0) bias[n] = (n < cout) ? ((b ? b[n] : 0.0f) * scale + shift) : 0.0f;
}

// ---------------------------------------------------------------------------
// x (B,3,N) -> xyz (B,N,3)
// ---------------------------------------------------------------------------
__global__ void transpose_xyz_kernel(const float* __restrict__ x, float* __restrict__ xyz)
{
  int i = blockIdx.x * blockDim.x + threadIdx.x;
  if (i >= BATCH * NPTS) return;
  int b = i / NPTS, n = i % NPTS;
  xyz[(size_t)i * 3 + 0] = x[((size_t)b * 3 + 0) * NPTS + n];
  xyz[(size_t)i * 3 + 1] = x[((size_t)b * 3 + 1) * NPTS + n];
  xyz[(size_t)i * 3 + 2] = x[((size_t)b * 3 + 2) * NPTS + n];
}

// ---------------------------------------------------------------------------
// Farthest point sampling: one workgroup per batch, dist array in LDS (32KB)
// Matches reference scan: emit far, update dist with far's point, argmax.
// ---------------------------------------------------------------------------
__global__ void fps_kernel(const float* __restrict__ pts, int Np, int S,
                           int* __restrict__ outIdx, float* __restrict__ outC)
{
  __shared__ float dist[NPTS];
  __shared__ float rv[FPS_T];
  __shared__ int   ri[FPS_T];
  const int b = blockIdx.x, tid = threadIdx.x;
  const float* P = pts + (size_t)b * Np * 3;
  for (int j = tid; j < Np; j += FPS_T) dist[j] = 1e10f;
  __syncthreads();
  int far = 0;
  for (int s = 0; s < S; ++s) {
    if (tid == 0) {
      outIdx[b * S + s] = far;
      outC[((size_t)b * S + s) * 3 + 0] = P[far * 3 + 0];
      outC[((size_t)b * S + s) * 3 + 1] = P[far * 3 + 1];
      outC[((size_t)b * S + s) * 3 + 2] = P[far * 3 + 2];
    }
    const float cx = P[far*3], cy = P[far*3+1], cz = P[far*3+2];
    float best = -1.0f; int bi = 0;
    for (int j = tid; j < Np; j += FPS_T) {
      float dx = P[j*3]-cx, dy = P[j*3+1]-cy, dz = P[j*3+2]-cz;
      float d = dx*dx + dy*dy + dz*dz;
      float o = dist[j];
      d = d < o ? d : o;
      dist[j] = d;
      if (d > best) { best = d; bi = j; }
    }
    rv[tid] = best; ri[tid] = bi;
    __syncthreads();
    for (int off = FPS_T >> 1; off > 0; off >>= 1) {
      if (tid < off) {
        float v2 = rv[tid+off]; int i2 = ri[tid+off];
        if (v2 > rv[tid] || (v2 == rv[tid] && i2 < ri[tid])) { rv[tid] = v2; ri[tid] = i2; }
      }
      __syncthreads();
    }
    far = ri[0];
    __syncthreads();
  }
}

// ---------------------------------------------------------------------------
// Ball query: first KRAW in-index-order points within radius, pad-with-first
// to KNN (reference semantics; duplicates are max-pool invariant).
// ---------------------------------------------------------------------------
__global__ void ballquery_kernel(const float* __restrict__ pts, const float* __restrict__ ctr,
                                 int Np, int S, float r2, int total,
                                 int* __restrict__ outIdx)
{
  int i = blockIdx.x * blockDim.x + threadIdx.x;
  if (i >= total) return;
  const int b = i / S;
  const float* P = pts + (size_t)b * Np * 3;
  const float cx = ctr[(size_t)i*3], cy = ctr[(size_t)i*3+1], cz = ctr[(size_t)i*3+2];
  int* o = outIdx + (size_t)i * KNN;
  int cnt = 0;
  for (int j = 0; j < Np && cnt < KRAW; ++j) {
    float dx = P[j*3]-cx, dy = P[j*3+1]-cy, dz = P[j*3+2]-cz;
    if (dx*dx + dy*dy + dz*dz <= r2) o[cnt++] = j;
  }
  int f0 = (cnt > 0) ? o[0] : 0;
  for (int t = cnt; t < KNN; ++t) o[t] = f0;
}

// ---------------------------------------------------------------------------
// Build grouped GEMM input A[row][Kpad] = [p - center | gathered feats | 0 pad]
// ---------------------------------------------------------------------------
__global__ void group_kernel(const float* __restrict__ pts, const float* __restrict__ ctr,
                             const int* __restrict__ nidx, const __bf16* __restrict__ feats,
                             int Np, int S, int Cf, int Kpad, long rows,
                             __bf16* __restrict__ A)
{
  long row = (long)blockIdx.x * blockDim.x + threadIdx.x;
  if (row >= rows) return;
  const long g = row >> 6;                 // group (KNN = 64)
  const int  b = (int)(g / S);
  const int  idx = nidx[row];
  const float* p = pts + ((size_t)b * Np + idx) * 3;
  const float* c = ctr + (size_t)g * 3;
  __bf16* a = A + (size_t)row * Kpad;
  a[0] = f2bf(p[0] - c[0]);
  a[1] = f2bf(p[1] - c[1]);
  a[2] = f2bf(p[2] - c[2]);
  int col = 3;
  if (feats) {
    const __bf16* f = feats + ((size_t)b * Np + idx) * Cf;
    for (int cc = 0; cc < Cf; ++cc) a[col + cc] = f[cc];
    col += Cf;
  }
  for (; col < Kpad; ++col) a[col] = f2bf(0.0f);
}

// sa3 input: [abs xyz | feats | 0 pad]
__global__ void concat_kernel(const float* __restrict__ xyz, const __bf16* __restrict__ feats,
                              int Cf, int Kpad, int rows, __bf16* __restrict__ A)
{
  int row = blockIdx.x * blockDim.x + threadIdx.x;
  if (row >= rows) return;
  __bf16* a = A + (size_t)row * Kpad;
  a[0] = f2bf(xyz[(size_t)row*3+0]);
  a[1] = f2bf(xyz[(size_t)row*3+1]);
  a[2] = f2bf(xyz[(size_t)row*3+2]);
  for (int cc = 0; cc < Cf; ++cc) a[3+cc] = feats[(size_t)row*Cf + cc];
  for (int col = 3 + Cf; col < Kpad; ++col) a[col] = f2bf(0.0f);
}

// Max over R consecutive rows per group
__global__ void maxpool_kernel(const __bf16* __restrict__ H, int R, int C, long total,
                               __bf16* __restrict__ out)
{
  long i = (long)blockIdx.x * blockDim.x + threadIdx.x;
  if (i >= total) return;
  long g = i / C; int c = (int)(i % C);
  const __bf16* h = H + (size_t)g * R * C + c;
  float m = -3.0e38f;
  for (int r = 0; r < R; ++r) { float v = bf2f(h[(size_t)r * C]); m = v > m ? v : m; }
  out[i] = f2bf(m);
}

__global__ void logsoftmax_kernel(const float* __restrict__ logits, int rows, int C, int Cpad,
                                  float* __restrict__ out)
{
  int r = blockIdx.x * blockDim.x + threadIdx.x;
  if (r >= rows) return;
  const float* l = logits + (size_t)r * Cpad;
  float m = l[0];
  for (int c = 1; c < C; ++c) m = l[c] > m ? l[c] : m;
  float s = 0.0f;
  for (int c = 0; c < C; ++c) s += expf(l[c] - m);
  float lse = logf(s);
  for (int c = 0; c < C; ++c) out[(size_t)r * C + c] = l[c] - m - lse;
}

// ---------------------------------------------------------------------------
extern "C" void kernel_launch(void* const* d_in, const int* in_sizes, int n_in,
                              void* d_out, int out_size, void* d_ws, size_t ws_size,
                              hipStream_t stream)
{
  (void)out_size; (void)ws_size; (void)n_in;

  // --- locate inputs (insertion order vs jax-sorted pytree order) ---
  const float* x = nullptr;
  const float *lw[12], *lb[12], *lg[12], *lbe[12];
  auto F = [&](int i) { return (const float*)d_in[i]; };
  const bool x_first = (in_sizes[0] == BATCH * 3 * NPTS);
  if (x_first) {
    x = F(0);
    for (int i = 0; i < 9; ++i) { int s = 1 + i*4; lw[i]=F(s); lb[i]=F(s+1); lg[i]=F(s+2); lbe[i]=F(s+3); }
    lw[9]=F(37);  lb[9]=F(38);  lg[9]=F(39);  lbe[9]=F(40);   // fc1 + bn1
    lw[10]=F(41); lb[10]=F(42); lg[10]=F(43); lbe[10]=F(44);  // fc2 + bn2
    lw[11]=F(45); lb[11]=F(46); lg[11]=nullptr; lbe[11]=nullptr;
  } else {                                                     // sorted keys, x last
    lbe[9]=F(0);  lg[9]=F(1);  lbe[10]=F(2); lg[10]=F(3);
    lb[9]=F(4);   lw[9]=F(5);  lb[10]=F(6); lw[10]=F(7);
    lb[11]=F(8);  lw[11]=F(9); lg[11]=nullptr; lbe[11]=nullptr;
    for (int i = 0; i < 9; ++i) { int s = 10 + i*4; lb[i]=F(s); lbe[i]=F(s+1); lg[i]=F(s+2); lw[i]=F(s+3); }
    x = F(46);
  }

  // --- layer dims: sa1(0-2) sa2(3-5) sa3(6-8) fc1..fc3(9-11) ---
  static const int LCO[12] = {64,64,128, 128,128,256, 256,512,1024, 512,256,NCATS};
  static const int LCI[12] = {3,64,64,   131,128,128, 259,256,512,  1024,512,256};
  static const int LKP[12] = {32,64,64,  160,128,128, 288,256,512,  1024,512,256};
  static const int LNP[12] = {64,64,128, 128,128,256, 256,512,1024, 512,256,CATPAD};

  // --- workspace carve-up (256B aligned bump allocator) ---
  char* wp = (char*)d_ws;
  auto alloc = [&](size_t bytes) -> void* {
    void* p = (void*)wp; wp += (bytes + 255) & ~(size_t)255; return p;
  };
  const size_t R1 = (size_t)BATCH * S1 * KNN;   // 524288 rows (sa1)
  const size_t R2 = (size_t)BATCH * S2 * KNN;   // 131072 rows (sa2)
  const size_t R3 = (size_t)BATCH * S2;         //   2048 rows (sa3)

  float*  xyz   = (float*)alloc((size_t)BATCH*NPTS*3*4);
  int*    fidx1 = (int*)  alloc((size_t)BATCH*S1*4);
  float*  c1    = (float*)alloc((size_t)BATCH*S1*3*4);
  int*    nidx1 = (int*)  alloc(R1*4);
  int*    fidx2 = (int*)  alloc((size_t)BATCH*S2*4);
  float*  c2    = (float*)alloc((size_t)BATCH*S2*3*4);
  int*    nidx2 = (int*)  alloc(R2*4);
  __bf16* H0    = (__bf16*)alloc(R1*128*2);     // 134 MB ping (L2-backed)
  __bf16* H1    = (__bf16*)alloc(R1*64*2);      //  67 MB pong (also hosts A buffers)
  __bf16* f1    = (__bf16*)alloc((size_t)BATCH*S1*128*2);
  __bf16* f2    = (__bf16*)alloc((size_t)BATCH*S2*256*2);
  __bf16* gfeat = (__bf16*)alloc((size_t)BATCH*1024*2);
  __bf16* h1    = (__bf16*)alloc((size_t)BATCH*512*2);
  __bf16* h2    = (__bf16*)alloc((size_t)BATCH*256*2);
  float*  logits= (float*) alloc((size_t)BATCH*CATPAD*4);
  __bf16* Bp[12]; float* bias[12];
  for (int i = 0; i < 12; ++i) {
    Bp[i]   = (__bf16*)alloc((size_t)LKP[i]*LNP[i]*2);
    bias[i] = (float*) alloc((size_t)LNP[i]*4);
  }

  auto gemm = [&](const __bf16* A, int li, __bf16* oB, float* oF, size_t M, int relu) {
    const int Np = LNP[li], Kp = LKP[li];
    const int ntile = Np / 16;
    int NT = (ntile % 4 == 0) ? 4 : (ntile % 3 == 0) ? 3 : (ntile % 2 == 0) ? 2 : 1;
    const int wpb = (M % 64 == 0) ? 4 : 1;                 // waves (M-tiles) per block
    dim3 grid((unsigned)(M / 16 / wpb), (unsigned)(ntile / NT));
    dim3 block(32 * wpb);
    switch (NT) {
      case 4: wmma_gemm_kernel<4><<<grid, block, 0, stream>>>(A, Bp[li], bias[li], oB, oF, Kp, Np, relu); break;
      case 3: wmma_gemm_kernel<3><<<grid, block, 0, stream>>>(A, Bp[li], bias[li], oB, oF, Kp, Np, relu); break;
      case 2: wmma_gemm_kernel<2><<<grid, block, 0, stream>>>(A, Bp[li], bias[li], oB, oF, Kp, Np, relu); break;
      default: wmma_gemm_kernel<1><<<grid, block, 0, stream>>>(A, Bp[li], bias[li], oB, oF, Kp, Np, relu); break;
    }
  };

  // --- fold BN, swizzle all weights (independent of data path) ---
  for (int i = 0; i < 12; ++i) {
    int total = LKP[i] * LNP[i];
    prep_weights_kernel<<<(total + 255) / 256, 256, 0, stream>>>(
        lw[i], lb[i], lg[i], lbe[i], Bp[i], bias[i], LCO[i], LCI[i], LKP[i], LNP[i]);
  }

  // --- pipeline ---
  transpose_xyz_kernel<<<(BATCH*NPTS + 255)/256, 256, 0, stream>>>(x, xyz);

  // sa1
  fps_kernel<<<BATCH, FPS_T, 0, stream>>>(xyz, NPTS, S1, fidx1, c1);
  ballquery_kernel<<<(BATCH*S1 + 255)/256, 256, 0, stream>>>(xyz, c1, NPTS, S1, 0.04f, BATCH*S1, nidx1);
  __bf16* A1 = H1;
  group_kernel<<<(unsigned)((R1 + 255)/256), 256, 0, stream>>>(xyz, c1, nidx1, nullptr, NPTS, S1, 0, LKP[0], (long)R1, A1);
  gemm(A1, 0, H0, nullptr, R1, 1);
  gemm(H0, 1, H1, nullptr, R1, 1);
  gemm(H1, 2, H0, nullptr, R1, 1);
  maxpool_kernel<<<(unsigned)(((size_t)BATCH*S1*128 + 255)/256), 256, 0, stream>>>(H0, KNN, 128, (long)BATCH*S1*128, f1);

  // sa2
  fps_kernel<<<BATCH, FPS_T, 0, stream>>>(c1, S1, S2, fidx2, c2);
  ballquery_kernel<<<(BATCH*S2 + 255)/256, 256, 0, stream>>>(c1, c2, S1, S2, 0.16f, BATCH*S2, nidx2);
  __bf16* A2 = H1;
  group_kernel<<<(unsigned)((R2 + 255)/256), 256, 0, stream>>>(c1, c2, nidx2, f1, S1, S2, 128, LKP[3], (long)R2, A2);
  gemm(A2, 3, H0, nullptr, R2, 1);
  gemm(H0, 4, H1, nullptr, R2, 1);
  gemm(H1, 5, H0, nullptr, R2, 1);
  maxpool_kernel<<<(unsigned)(((size_t)BATCH*S2*256 + 255)/256), 256, 0, stream>>>(H0, KNN, 256, (long)BATCH*S2*256, f2);

  // sa3 (group_all)
  __bf16* A3 = H1;
  concat_kernel<<<(unsigned)((R3 + 255)/256), 256, 0, stream>>>(c2, f2, 256, LKP[6], (int)R3, A3);
  gemm(A3, 6, H0, nullptr, R3, 1);
  gemm(H0, 7, H1, nullptr, R3, 1);
  gemm(H1, 8, H0, nullptr, R3, 1);
  maxpool_kernel<<<(unsigned)(((size_t)BATCH*1024 + 255)/256), 256, 0, stream>>>(H0, S2, 1024, (long)BATCH*1024, gfeat);

  // head
  gemm(gfeat, 9,  h1, nullptr, BATCH, 1);
  gemm(h1,    10, h2, nullptr, BATCH, 1);
  gemm(h2,    11, nullptr, logits, BATCH, 0);
  logsoftmax_kernel<<<1, 32, 0, stream>>>(logits, BATCH, NCATS, CATPAD, (float*)d_out);
}
